// DETR_7756710937315
// MI455X (gfx1250) — compile-verified, matrix operand order checked by reading, not run on hardware
//
#include <hip/hip_runtime.h>
#include <hip/hip_bf16.h>
#include <math.h>

typedef __attribute__((ext_vector_type(16))) _Float16 v16h;
typedef __attribute__((ext_vector_type(8)))  float    v8f;

#define BN_INV 0.9999950000374997f
#define LSEQ 49
#define NHEAD 8
#define HD 32

// ---------------------------------------------------------------------------
// Pack a row-major f32 matrix [M,K] (row stride lda) into f16 WMMA operand
// layout: tile (mt,kt) of 16x32 stored as 32 lanes x 16 halves, matching the
// CDNA5 16-bit A-operand layout (lanes 0-15: K 0-7/16-23 of rows 0-15,
// lanes 16-31: K 8-15/24-31).  B operands (W[N,K], lane = output column) use
// the identical packing.  Output tile-major: [mt][kt][lane][16].
// ---------------------------------------------------------------------------
__global__ void k_pack(const float* __restrict__ A, int lda,
                       _Float16* __restrict__ out, int KT, int total)
{
    int o = blockIdx.x * blockDim.x + threadIdx.x;
    if (o >= total) return;
    int idx  = o & 15;
    int lane = (o >> 4) & 31;
    int tile = o >> 9;
    int kt = tile % KT;
    int mt = tile / KT;
    int half = lane >> 4;
    int mrow = lane & 15;
    int kk = (idx & 7) + ((idx & 8) ? 16 : 0) + (half ? 8 : 0);
    out[o] = (_Float16)A[(size_t)(mt * 16 + mrow) * lda + kt * 32 + kk];
}

// ---------------------------------------------------------------------------
// WMMA GEMM on pre-packed f16 operands.
// Wave computes a 32x32 C tile (2 A tiles x 2 B tiles -> 4 accs, 4 WMMA/step).
// Block = 128 threads = 4 waves in a 2x2 arrangement -> 64x64 C per block.
// M%64==0, N%64==0, K%32==0.
// epilogue mode: 0 = +bias, 1 = relu(+bias), 2 = lrelu(BN_INV*g*acc + b)
// ---------------------------------------------------------------------------
__device__ __forceinline__ void epi_store(const v8f& acc, int mb, int nb,
                                          int half, int lr,
                                          const float* __restrict__ bias,
                                          const float* __restrict__ gam,
                                          const float* __restrict__ bet,
                                          int mode, float* __restrict__ C,
                                          int ldc, int coff)
{
    const int n = nb + lr;
    float scale = 1.0f, shift = 0.0f;
    if (mode == 2) { scale = BN_INV * gam[n]; shift = bet[n]; }
    else if (bias) { shift = bias[n]; }
#pragma unroll
    for (int r = 0; r < 8; ++r) {
        const int m = mb + r + half * 8;
        float v = fmaf(acc[r], scale, shift);
        if (mode == 1)      v = fmaxf(v, 0.0f);
        else if (mode == 2) v = (v > 0.0f) ? v : 0.1f * v;
        C[(size_t)m * ldc + coff + n] = v;
    }
}

__global__ void k_gemm2(const _Float16* __restrict__ Apk,
                        const _Float16* __restrict__ Bpk,
                        const float* __restrict__ bias,
                        const float* __restrict__ gam,
                        const float* __restrict__ bet,
                        int mode,
                        float* __restrict__ C, int ldc, int coff, int KT)
{
    const int lane = threadIdx.x & 31;
    const int wave = threadIdx.x >> 5;
    const int wm = wave & 1, wn = wave >> 1;
    const int mt0 = blockIdx.y * 4 + wm * 2;   // 16-row tiles
    const int nt0 = blockIdx.x * 4 + wn * 2;   // 16-col tiles

    const _Float16* ap0 = Apk + ((size_t)mt0 * KT) * 512 + lane * 16;
    const _Float16* ap1 = ap0 + (size_t)KT * 512;
    const _Float16* bp0 = Bpk + ((size_t)nt0 * KT) * 512 + lane * 16;
    const _Float16* bp1 = bp0 + (size_t)KT * 512;

    v8f acc00, acc01, acc10, acc11;
#pragma unroll
    for (int r = 0; r < 8; ++r) { acc00[r] = 0.f; acc01[r] = 0.f; acc10[r] = 0.f; acc11[r] = 0.f; }

    for (int kt = 0; kt < KT; ++kt) {
        const size_t off = (size_t)kt * 512;
        __builtin_prefetch(ap0 + off + 4 * 512, 0, 1);   // global_prefetch_b8
        __builtin_prefetch(bp0 + off + 4 * 512, 0, 1);
        v16h a0 = *(const v16h*)(ap0 + off);
        v16h a1 = *(const v16h*)(ap1 + off);
        v16h b0 = *(const v16h*)(bp0 + off);
        v16h b1 = *(const v16h*)(bp1 + off);
        acc00 = __builtin_amdgcn_wmma_f32_16x16x32_f16(false, a0, false, b0, (short)0, acc00, false, false);
        acc01 = __builtin_amdgcn_wmma_f32_16x16x32_f16(false, a0, false, b1, (short)0, acc01, false, false);
        acc10 = __builtin_amdgcn_wmma_f32_16x16x32_f16(false, a1, false, b0, (short)0, acc10, false, false);
        acc11 = __builtin_amdgcn_wmma_f32_16x16x32_f16(false, a1, false, b1, (short)0, acc11, false, false);
    }

    const int half = lane >> 4, lr = lane & 15;
    const int mb = mt0 * 16, nb = nt0 * 16;
    epi_store(acc00, mb,      nb,      half, lr, bias, gam, bet, mode, C, ldc, coff);
    epi_store(acc01, mb,      nb + 16, half, lr, bias, gam, bet, mode, C, ldc, coff);
    epi_store(acc10, mb + 16, nb,      half, lr, bias, gam, bet, mode, C, ldc, coff);
    epi_store(acc11, mb + 16, nb + 16, half, lr, bias, gam, bet, mode, C, ldc, coff);
}

// ---------------------------------------------------------------------------
// Scalar softmax-attention: one thread per (b, h, q).  Lq = Lk = 49 fixed.
// ---------------------------------------------------------------------------
__global__ void k_attn(const float* __restrict__ Q, int qld,
                       const float* __restrict__ K, int kld,
                       const float* __restrict__ V, int vld,
                       float* __restrict__ O, float scale, int total)
{
    int t = blockIdx.x * blockDim.x + threadIdx.x;
    if (t >= total) return;
    int q = t % LSEQ;
    int h = (t / LSEQ) % NHEAD;
    int b = t / (LSEQ * NHEAD);

    const float* qp = Q + (size_t)(b * LSEQ + q) * qld + h * HD;
    const float* kp = K + (size_t)(b * LSEQ) * kld + h * HD;
    const float* vp = V + (size_t)(b * LSEQ) * vld + h * HD;

    float qv[HD];
#pragma unroll
    for (int d = 0; d < HD; ++d) qv[d] = qp[d];

    float sc[LSEQ];
    float mx = -1e30f;
#pragma unroll 1
    for (int k = 0; k < LSEQ; ++k) {
        float s = 0.0f;
        const float* kr = kp + (size_t)k * kld;
#pragma unroll
        for (int d = 0; d < HD; ++d) s = fmaf(qv[d], kr[d], s);
        s *= scale;
        sc[k] = s;
        mx = fmaxf(mx, s);
    }
    float sum = 0.0f;
#pragma unroll 1
    for (int k = 0; k < LSEQ; ++k) { float e = __expf(sc[k] - mx); sc[k] = e; sum += e; }
    float inv = 1.0f / sum;

    float out[HD];
#pragma unroll
    for (int d = 0; d < HD; ++d) out[d] = 0.0f;
#pragma unroll 1
    for (int k = 0; k < LSEQ; ++k) {
        float w = sc[k] * inv;
        const float* vr = vp + (size_t)k * vld;
#pragma unroll
        for (int d = 0; d < HD; ++d) out[d] = fmaf(w, vr[d], out[d]);
    }
    float* op = O + (size_t)(b * LSEQ + q) * 256 + h * HD;
#pragma unroll
    for (int d = 0; d < HD; ++d) op[d] = out[d];
}

// ---------------------------------------------------------------------------
// LayerNorm over D=256 with optional residual; one 256-thread block per row.
// ---------------------------------------------------------------------------
__global__ void k_add_ln(const float* __restrict__ x, const float* __restrict__ res,
                         const float* __restrict__ g, const float* __restrict__ b,
                         float* __restrict__ out)
{
    __shared__ float red[256];
    const int i = threadIdx.x;
    const size_t row = blockIdx.x;
    float v = x[row * 256 + i];
    if (res) v += res[row * 256 + i];
    red[i] = v;
    __syncthreads();
    for (int s = 128; s > 0; s >>= 1) { if (i < s) red[i] += red[i + s]; __syncthreads(); }
    const float mean = red[0] * (1.0f / 256.0f);
    __syncthreads();
    float d = v - mean;
    red[i] = d * d;
    __syncthreads();
    for (int s = 128; s > 0; s >>= 1) { if (i < s) red[i] += red[i + s]; __syncthreads(); }
    const float rstd = rsqrtf(red[0] * (1.0f / 256.0f) + 1e-5f);
    out[row * 256 + i] = d * rstd * g[i] + b[i];
}

// ---------------------------------------------------------------------------
// Misc layout kernels
// ---------------------------------------------------------------------------
__global__ void k_transpose_src(const float* __restrict__ src, float* __restrict__ X, int total)
{   // X[(b*49+s)*256 + d] = src[((s*64)+b)*256 + d]
    int i = blockIdx.x * blockDim.x + threadIdx.x;
    if (i >= total) return;
    int d = i & 255; int r = i >> 8;
    int b = r / LSEQ, s = r % LSEQ;
    X[i] = src[((size_t)s * 64 + b) * 256 + d];
}

__global__ void k_bcast_query(const float* __restrict__ qy, float* __restrict__ T, int total)
{   // T[(b*49+q)*256 + d] = query[q*256 + d]
    int i = blockIdx.x * blockDim.x + threadIdx.x;
    if (i >= total) return;
    int d = i & 255; int r = i >> 8;
    int q = r % LSEQ;
    T[i] = qy[(size_t)q * 256 + d];
}

__global__ void k_upsample(const float* __restrict__ T, float* __restrict__ fused,
                           int F, int f, int total)
{   // fused[(b*F*F + y*F + x)*512 + d] = T[(b*49 + (y/f)*7 + (x/f))*256 + d]
    int i = blockIdx.x * blockDim.x + threadIdx.x;
    if (i >= total) return;
    int d = i & 255; int r = i >> 8;
    int FF = F * F;
    int p = r % FF, b = r / FF;
    int y = p / F, x = p % F;
    int q = (y / f) * 7 + (x / f);
    fused[(size_t)r * 512 + d] = T[((size_t)b * LSEQ + q) * 256 + d];
}

__global__ void k_gather_spa(const float* __restrict__ sf, float* __restrict__ SPA,
                             int Kd, int FF, int total)
{   // SPA[m*K + k] = sf[b*(K*FF) + k*FF + p],  m = b*FF + p, k = c*T + t
    int i = blockIdx.x * blockDim.x + threadIdx.x;
    if (i >= total) return;
    int k = i % Kd; int m = i / Kd;
    int p = m % FF, b = m / FF;
    SPA[(size_t)m * Kd + k] = sf[(size_t)b * Kd * FF + (size_t)k * FF + p];
}

// Depthwise 3x3 (pad 1) over channels-last [B,F,F,64] + BN + LeakyReLU
__global__ void k_dw(const float* __restrict__ in, const float* __restrict__ w,
                     const float* __restrict__ g, const float* __restrict__ bb,
                     float* __restrict__ out, int F, int total)
{
    int i = blockIdx.x * blockDim.x + threadIdx.x;
    if (i >= total) return;
    int c = i & 63; int r = i >> 6;
    int FF = F * F;
    int p = r % FF, b = r / FF;
    int y = p / F, x = p % F;
    float sum = 0.0f;
#pragma unroll
    for (int ky = 0; ky < 3; ++ky) {
#pragma unroll
        for (int kx = 0; kx < 3; ++kx) {
            int yy = y + ky - 1, xx = x + kx - 1;
            if (yy >= 0 && yy < F && xx >= 0 && xx < F) {
                sum = fmaf(in[((size_t)b * FF + yy * F + xx) * 64 + c],
                           w[c * 9 + ky * 3 + kx], sum);
            }
        }
    }
    float v = fmaf(BN_INV * g[c], sum, bb[c]);
    out[i] = (v > 0.0f) ? v : 0.1f * v;
}

// Final predictions + box decode, written directly into out[B,1029,85]
__global__ void k_final(const float* __restrict__ CF, const float* __restrict__ RF,
                        const float* __restrict__ conf_w, const float* __restrict__ conf_b,
                        const float* __restrict__ clsp_w, const float* __restrict__ clsp_b,
                        const float* __restrict__ regp_w, const float* __restrict__ regp_b,
                        float* __restrict__ out, int F, float s, int aoff, int total)
{
    int i = blockIdx.x * blockDim.x + threadIdx.x;
    if (i >= total) return;
    int c = i % 85; int m = i / 85;
    int FF = F * F;
    int p = m % FF, b = m / FF;
    float* o = out + ((size_t)b * 1029 + aoff + p) * 85 + c;
    if (c == 0) {
        const float* rf = RF + (size_t)m * 64;
        float v = conf_b[0];
#pragma unroll
        for (int d = 0; d < 64; ++d) v = fmaf(rf[d], conf_w[d], v);
        *o = v;
    } else if (c < 81) {
        const float* cf = CF + (size_t)m * 64;
        const float* w = clsp_w + (size_t)(c - 1) * 64;
        float v = clsp_b[c - 1];
#pragma unroll
        for (int d = 0; d < 64; ++d) v = fmaf(cf[d], w[d], v);
        *o = v;
    } else {
        const float* rf = RF + (size_t)m * 64;
        float rg[4];
#pragma unroll
        for (int j = 0; j < 4; ++j) {
            float v = regp_b[j];
            const float* w = regp_w + j * 64;
#pragma unroll
            for (int d = 0; d < 64; ++d) v = fmaf(rf[d], w[d], v);
            rg[j] = v;
        }
        float ax = ((float)(p % F) + 0.5f) * s;
        float ay = ((float)(p / F) + 0.5f) * s;
        float cx = ax + rg[0] * s;
        float cy = ay + rg[1] * s;
        float bw = __expf(rg[2]) * s;
        float bh = __expf(rg[3]) * s;
        float v;
        if      (c == 81) v = cx - 0.5f * bw;
        else if (c == 82) v = cy - 0.5f * bh;
        else if (c == 83) v = cx + 0.5f * bw;
        else              v = cy + 0.5f * bh;
        *o = v;
    }
}

// ---------------------------------------------------------------------------
// Host orchestration
// ---------------------------------------------------------------------------
#define DEC(l, o)  (4 + (l) * 18 + (o))
#define DECNORM_B 40
#define DECNORM_G 41
#define ENC(l, o)  (42 + (l) * 12 + (o))
#define ENCNORM_B 66
#define ENCNORM_G 67
#define LVL(l, o)  (68 + (l) * 30 + (o))
#define QUERY_IDX 158

struct GemmCtx {
    hipStream_t st;
    _Float16* APK;
    _Float16* WPK;
};

static inline void pack(const GemmCtx& cx, const float* A, int lda, _Float16* dst, int M, int K)
{
    int total = M * K;
    hipLaunchKernelGGL(k_pack, dim3((total + 255) / 256), dim3(256), 0, cx.st,
                       A, lda, dst, K / 32, total);
}

static inline void gemm2(const GemmCtx& cx, const float* bias, const float* gam,
                         const float* bet, int mode, float* C, int ldc, int coff,
                         int M, int N, int K)
{
    dim3 g(N / 64, M / 64), b(128);
    hipLaunchKernelGGL(k_gemm2, g, b, 0, cx.st, cx.APK, cx.WPK,
                       bias, gam, bet, mode, C, ldc, coff, K / 32);
}

// pack A + pack W + GEMM
static inline void gemm_full(const GemmCtx& cx, const float* A, int lda, const float* W,
                             const float* bias, const float* gam, const float* bet,
                             int mode, float* C, int ldc, int coff, int M, int N, int K)
{
    pack(cx, A, lda, cx.APK, M, K);
    pack(cx, W, K, cx.WPK, N, K);
    gemm2(cx, bias, gam, bet, mode, C, ldc, coff, M, N, K);
}

extern "C" void kernel_launch(void* const* d_in, const int* in_sizes, int n_in,
                              void* d_out, int out_size, void* d_ws, size_t ws_size,
                              hipStream_t stream)
{
    (void)in_sizes; (void)n_in; (void)out_size; (void)ws_size;
    const float* src = (const float*)d_in[0];
    const float* sf[3] = { (const float*)d_in[1], (const float*)d_in[2], (const float*)d_in[3] };
    auto P = [&](int i) -> const float* { return (const float*)d_in[i]; };
    float* ws = (float*)d_ws;
    float* out = (float*)d_out;

    const int M = 64 * LSEQ;               // 3136 tokens
    const float att_scale = 0.17677669529663689f;  // 32^-0.5

    // workspace layout (floats)
    float* T    = ws;                       //  3136 x 256 (persists for heads)
    float* X    = ws + 802816;              //  3136 x 256
    float* QKV  = ws + 1605632;             //  3136 x 768
    float* O    = ws + 4014080;             //  3136 x 256
    float* H    = ws + 4816896;             //  3136 x 2048
    float* MEM  = ws + 11239424;            //  3136 x 256
    float* Q2   = ws + 12042240;            //  3136 x 256
    float* KV   = ws + 12845056;            //  3136 x 512
    float* TMP  = ws + 14450688;            //  3136 x 256
    // head region (reuses transformer scratch, keeps T)
    float* FUSED = X;                       // up to 50176 x 512
    float* SPA   = ws + 26492928;           // up to 50176 x 128
    float* F1    = ws + 32915456;           // up to 50176 x 256
    float* CF    = ws + 45760512;           // up to 50176 x 64
    float* CF2   = ws + 48971776;
    float* RF    = ws + 52183040;
    float* RF2   = ws + 55394304;
    // packed f16 operand panels
    GemmCtx cx;
    cx.st  = stream;
    cx.APK = (_Float16*)(ws + 58605568);    // up to 50176 x 512 halves
    cx.WPK = (_Float16*)(ws + 71450624);    // up to 2048 x 256 halves

    dim3 b256(256);
    auto gr = [&](int n) { return dim3((n + 255) / 256); };

    // ---- encoder ----
    hipLaunchKernelGGL(k_transpose_src, gr(M * 256), b256, 0, stream, src, X, M * 256);
    for (int l = 0; l < 2; ++l) {
        gemm_full(cx, X, 256, P(ENC(l,7)), P(ENC(l,6)), nullptr, nullptr, 0, QKV, 768, 0, M, 768, 256);
        hipLaunchKernelGGL(k_attn, gr(M * NHEAD), b256, 0, stream,
                           QKV, 768, QKV + 256, 768, QKV + 512, 768, O, att_scale, M * NHEAD);
        gemm_full(cx, O, 256, P(ENC(l,9)), P(ENC(l,8)), nullptr, nullptr, 0, TMP, 256, 0, M, 256, 256);
        hipLaunchKernelGGL(k_add_ln, dim3(M), b256, 0, stream, X, TMP, P(ENC(l,3)), P(ENC(l,2)), X);
        gemm_full(cx, X, 256, P(ENC(l,10)), P(ENC(l,0)), nullptr, nullptr, 1, H, 2048, 0, M, 2048, 256);
        gemm_full(cx, H, 2048, P(ENC(l,11)), P(ENC(l,1)), nullptr, nullptr, 0, TMP, 256, 0, M, 256, 2048);
        hipLaunchKernelGGL(k_add_ln, dim3(M), b256, 0, stream, X, TMP, P(ENC(l,5)), P(ENC(l,4)), X);
    }
    hipLaunchKernelGGL(k_add_ln, dim3(M), b256, 0, stream, X, (const float*)nullptr,
                       P(ENCNORM_G), P(ENCNORM_B), MEM);

    // ---- decoder ----
    hipLaunchKernelGGL(k_bcast_query, gr(M * 256), b256, 0, stream, P(QUERY_IDX), T, M * 256);
    for (int l = 0; l < 2; ++l) {
        // self-attention
        gemm_full(cx, T, 256, P(DEC(l,13)), P(DEC(l,12)), nullptr, nullptr, 0, QKV, 768, 0, M, 768, 256);
        hipLaunchKernelGGL(k_attn, gr(M * NHEAD), b256, 0, stream,
                           QKV, 768, QKV + 256, 768, QKV + 512, 768, O, att_scale, M * NHEAD);
        gemm_full(cx, O, 256, P(DEC(l,15)), P(DEC(l,14)), nullptr, nullptr, 0, TMP, 256, 0, M, 256, 256);
        hipLaunchKernelGGL(k_add_ln, dim3(M), b256, 0, stream, T, TMP, P(DEC(l,7)), P(DEC(l,6)), T);
        // cross-attention (Q from t; K/V from mem via row split of in_w)
        gemm_full(cx, T, 256, P(DEC(l,3)), P(DEC(l,2)), nullptr, nullptr, 0, Q2, 256, 0, M, 256, 256);
        gemm_full(cx, MEM, 256, P(DEC(l,3)) + 256 * 256, P(DEC(l,2)) + 256, nullptr, nullptr, 0,
                  KV, 512, 0, M, 512, 256);
        hipLaunchKernelGGL(k_attn, gr(M * NHEAD), b256, 0, stream,
                           Q2, 256, KV, 512, KV + 256, 512, O, att_scale, M * NHEAD);
        gemm_full(cx, O, 256, P(DEC(l,5)), P(DEC(l,4)), nullptr, nullptr, 0, TMP, 256, 0, M, 256, 256);
        hipLaunchKernelGGL(k_add_ln, dim3(M), b256, 0, stream, T, TMP, P(DEC(l,9)), P(DEC(l,8)), T);
        // FFN
        gemm_full(cx, T, 256, P(DEC(l,16)), P(DEC(l,0)), nullptr, nullptr, 1, H, 2048, 0, M, 2048, 256);
        gemm_full(cx, H, 2048, P(DEC(l,17)), P(DEC(l,1)), nullptr, nullptr, 0, TMP, 256, 0, M, 256, 2048);
        hipLaunchKernelGGL(k_add_ln, dim3(M), b256, 0, stream, T, TMP, P(DEC(l,11)), P(DEC(l,10)), T);
    }
    hipLaunchKernelGGL(k_add_ln, dim3(M), b256, 0, stream, T, (const float*)nullptr,
                       P(DECNORM_G), P(DECNORM_B), T);

    // ---- detection heads ----
    const int   FS[3]   = { 28, 14, 7 };
    const int   UP[3]   = { 4, 2, 1 };
    const int   K1[3]   = { 32 * 4, 96 * 2, 1280 * 1 };
    const float ST[3]   = { 8.0f, 16.0f, 32.0f };
    const int   AOFF[3] = { 0, 784, 980 };

    for (int lvl = 0; lvl < 3; ++lvl) {
        const int F  = FS[lvl];
        const int FF = F * F;
        const int Mh = 64 * FF;
        const int Kd = K1[lvl];
        // upsampled hs -> fused[:, 0:256]
        hipLaunchKernelGGL(k_upsample, gr(Mh * 256), b256, 0, stream, T, FUSED, F, UP[lvl], Mh * 256);
        // gather spatial features channels-last
        hipLaunchKernelGGL(k_gather_spa, gr(Mh * Kd), b256, 0, stream, sf[lvl], SPA, Kd, FF, Mh * Kd);
        // dc1 (conv3d == GEMM over C*T) + BN + lrelu -> fused[:, 256:512]
        gemm_full(cx, SPA, Kd, P(LVL(lvl,15)), nullptr, P(LVL(lvl,14)), P(LVL(lvl,13)), 2,
                  FUSED, 512, 256, Mh, 256, Kd);
        // dc2: 512 -> 256
        gemm_full(cx, FUSED, 512, P(LVL(lvl,18)), nullptr, P(LVL(lvl,17)), P(LVL(lvl,16)), 2,
                  F1, 256, 0, Mh, 256, 512);
        // branch convs 256 -> 64 (pack F1 once, swap W panels)
        pack(cx, F1, 256, cx.APK, Mh, 256);
        pack(cx, P(LVL(lvl,2)), 256, cx.WPK, 64, 256);
        gemm2(cx, nullptr, P(LVL(lvl,1)),  P(LVL(lvl,0)),  2, CF, 64, 0, Mh, 64, 256);
        pack(cx, P(LVL(lvl,21)), 256, cx.WPK, 64, 256);
        gemm2(cx, nullptr, P(LVL(lvl,20)), P(LVL(lvl,19)), 2, RF, 64, 0, Mh, 64, 256);
        // depthwise 3x3 + BN + lrelu
        hipLaunchKernelGGL(k_dw, gr(Mh * 64), b256, 0, stream, CF, P(LVL(lvl,5)),
                           P(LVL(lvl,4)), P(LVL(lvl,3)), CF2, F, Mh * 64);
        hipLaunchKernelGGL(k_dw, gr(Mh * 64), b256, 0, stream, RF, P(LVL(lvl,24)),
                           P(LVL(lvl,23)), P(LVL(lvl,22)), RF2, F, Mh * 64);
        // pointwise 64 -> 64
        gemm_full(cx, CF2, 64, P(LVL(lvl,8)),  nullptr, P(LVL(lvl,7)),  P(LVL(lvl,6)),  2, CF, 64, 0, Mh, 64, 64);
        gemm_full(cx, RF2, 64, P(LVL(lvl,27)), nullptr, P(LVL(lvl,26)), P(LVL(lvl,25)), 2, RF, 64, 0, Mh, 64, 64);
        // predictions + box decode into out[B,1029,85]
        hipLaunchKernelGGL(k_final, gr(Mh * 85), b256, 0, stream, CF, RF,
                           P(LVL(lvl,12)), P(LVL(lvl,11)), P(LVL(lvl,10)), P(LVL(lvl,9)),
                           P(LVL(lvl,29)), P(LVL(lvl,28)), out, F, ST[lvl], AOFF[lvl], Mh * 85);
    }
}
